// Actor_Critic_40638980554875
// MI455X (gfx1250) — matrix operand order repaired and checked by reading, not compiled, not run on hardware
//
#include <hip/hip_runtime.h>
#include <hip/hip_bf16.h>
#include <cstdint>
#include <cstddef>

// ---------------------------------------------------------------------------
// Actor-Critic forward on MI455X (gfx1250), wave32 + WMMA f32_16x16x32_f16.
// All GEMM-shaped work runs through v_wmma_f32_16x16x32_f16 (f16 in, f32 acc)
// with 2x2 register blocking (32x32 output per wave) so each A/B fragment is
// reused twice: 4 WMMAs per 8 b128 loads instead of 4 per 16.
// Elementwise GRU gates / attention / sampling are small VALU kernels.
// Sampling: deterministic hash-based Gumbel-max (JAX threefry not replicated).
// Workspace use ~70 MB (assumed available in d_ws).
// ---------------------------------------------------------------------------

typedef __attribute__((ext_vector_type(16))) _Float16 v16h;
typedef __attribute__((ext_vector_type(8)))  _Float16 v8h;
typedef __attribute__((ext_vector_type(8)))  float    v8f;

#define B_    256
#define LSEQ  33
#define HID_  256
#define RNNIN 128
#define G3    768
#define WD_   128
#define MAXS_ 16

// ----------------------------- WMMA helpers --------------------------------
// A fragment, 16x32 f16 (M x K): lanes 0-15 rows M=0..15 hold K 0..15,
// lanes 16-31 same rows hold K 16..31. Two contiguous 8-half chunks.
__device__ __forceinline__ v16h load_fragA_ptr(const _Float16* rowptr, int kb, int lane) {
    const _Float16* p = rowptr + kb + ((lane >> 4) << 4);
    v8h c0 = *(const v8h*)(p);
    v8h c1 = *(const v8h*)(p + 8);
    v16h r;
#pragma unroll
    for (int i = 0; i < 8; ++i) { r[i] = c0[i]; r[8 + i] = c1[i]; }
    return r;
}
__device__ __forceinline__ v16h load_fragA_rm(const _Float16* A, int lda, int m0, int kb, int lane) {
    return load_fragA_ptr(A + (size_t)(m0 + (lane & 15)) * lda, kb, lane);
}
// B fragment, 32x16 f16 (K x N), weights stored row-major [N, K]:
// lanes 0-15 column N=lane hold K 0..15 contiguous, lanes 16-31 hold K 16..31.
__device__ __forceinline__ v16h load_fragB(const _Float16* W, int ldw, int n0, int kb, int lane) {
    const _Float16* p = W + (size_t)(n0 + (lane & 15)) * ldw + kb + ((lane >> 4) << 4);
    return *(const v16h*)p;
}
__device__ __forceinline__ v8f wmma32(v16h a, v16h b, v8f c) {
    return __builtin_amdgcn_wmma_f32_16x16x32_f16(false, a, false, b, (short)0, c, false, false);
}
// C/D layout: col = lane&15, rows = (lane>>4)*8 + i.
__device__ __forceinline__ void store_tile_f32(float* C, int ldc, int m0, int n0,
                                               v8f acc, int lane, float bv, int relu) {
    const int col = lane & 15, r0 = (lane >> 4) << 3;
#pragma unroll
    for (int i = 0; i < 8; ++i) {
        float v = acc[i] + bv;
        if (relu) v = fmaxf(v, 0.0f);
        C[(size_t)(m0 + r0 + i) * ldc + n0 + col] = v;
    }
}

// ----------------------------- generic GEMM --------------------------------
// C[M,N] (f32) = A[M,K] (f16 row-major) x W[N,K]^T (f16) (+bias) (+relu)
// 2x2 register-blocked: each wave owns a 32x32 output macro-tile.
// Requires M % 32 == 0, N % 32 == 0, K % 32 == 0 (true for all call sites).
__global__ void gemm_f16f32(const _Float16* __restrict__ A, int lda,
                            const _Float16* __restrict__ W, int ldw,
                            float* __restrict__ C, int ldc,
                            int M, int N, int K,
                            const float* __restrict__ bias, int relu) {
    const int lane = threadIdx.x & 31;
    const int wpb  = blockDim.x >> 5;
    const int tn = N >> 5;
    const int tiles = (M >> 5) * tn;
    const int stride = gridDim.x * wpb;
    for (int t = blockIdx.x * wpb + (threadIdx.x >> 5); t < tiles; t += stride) {
        const int m0 = (t / tn) << 5, n0 = (t % tn) << 5;
        v8f a00 = {}, a01 = {}, a10 = {}, a11 = {};
        const _Float16* ar0 = A + (size_t)(m0 + (lane & 15)) * lda;
        const _Float16* ar1 = ar0 + (size_t)16 * lda;
        for (int kb = 0; kb < K; kb += 32) {
            if (kb + 32 < K) {
                __builtin_prefetch(ar0 + kb + 32, 0, 0);
                __builtin_prefetch(ar1 + kb + 32, 0, 0);
            }
            v16h fa0 = load_fragA_ptr(ar0, kb, lane);
            v16h fa1 = load_fragA_ptr(ar1, kb, lane);
            v16h fb0 = load_fragB(W, ldw, n0, kb, lane);
            v16h fb1 = load_fragB(W, ldw, n0 + 16, kb, lane);
            a00 = wmma32(fa0, fb0, a00);
            a01 = wmma32(fa0, fb1, a01);
            a10 = wmma32(fa1, fb0, a10);
            a11 = wmma32(fa1, fb1, a11);
        }
        const int col = lane & 15;
        const float b0 = bias ? bias[n0 + col] : 0.0f;
        const float b1 = bias ? bias[n0 + 16 + col] : 0.0f;
        store_tile_f32(C, ldc, m0,      n0,      a00, lane, b0, relu);
        store_tile_f32(C, ldc, m0,      n0 + 16, a01, lane, b1, relu);
        store_tile_f32(C, ldc, m0 + 16, n0,      a10, lane, b0, relu);
        store_tile_f32(C, ldc, m0 + 16, n0 + 16, a11, lane, b1, relu);
    }
}

// ------------------- res conv as GEMM (K=32 padded from 27) ----------------
// A[12544,32] x W[512,32]^T, relu+bias, scatter into padded channels-last
// feature map featcl[B,9,9,512] (interior 7x7 at offset +1,+1). 2x2 blocked.
__device__ __forceinline__ void store_res(_Float16* featcl, int m0, int n0, v8f acc,
                                          int lane, const float* resb) {
    const int col = lane & 15, r0 = (lane >> 4) << 3;
    const float bv = resb[n0 + col];
#pragma unroll
    for (int i = 0; i < 8; ++i) {
        const int r = m0 + r0 + i;
        const int b = r / 49, p = r % 49, oy = p / 7, ox = p % 7;
        featcl[((size_t)b * 81 + (oy + 1) * 9 + (ox + 1)) * 512 + n0 + col] =
            (_Float16)fmaxf(acc[i] + bv, 0.0f);
    }
}
__global__ void gemm_res(const _Float16* __restrict__ A, const _Float16* __restrict__ W,
                         const float* __restrict__ resb, _Float16* __restrict__ featcl) {
    const int lane = threadIdx.x & 31;
    const int wpb = blockDim.x >> 5;
    const int tn = 512 >> 5, tiles = (12544 >> 5) * tn;
    const int stride = gridDim.x * wpb;
    for (int t = blockIdx.x * wpb + (threadIdx.x >> 5); t < tiles; t += stride) {
        const int m0 = (t / tn) << 5, n0 = (t % tn) << 5;
        v16h fa0 = load_fragA_rm(A, 32, m0, 0, lane);
        v16h fa1 = load_fragA_rm(A, 32, m0 + 16, 0, lane);
        v16h fb0 = load_fragB(W, 32, n0, 0, lane);
        v16h fb1 = load_fragB(W, 32, n0 + 16, 0, lane);
        v8f a00 = {}, a01 = {}, a10 = {}, a11 = {};
        a00 = wmma32(fa0, fb0, a00);
        a01 = wmma32(fa0, fb1, a01);
        a10 = wmma32(fa1, fb0, a10);
        a11 = wmma32(fa1, fb1, a11);
        store_res(featcl, m0,      n0,      a00, lane, resb);
        store_res(featcl, m0,      n0 + 16, a01, lane, resb);
        store_res(featcl, m0 + 16, n0,      a10, lane, resb);
        store_res(featcl, m0 + 16, n0 + 16, a11, lane, resb);
    }
}

// ---------------- fin conv: 9 shifted WMMA GEMMs over IC=512 ---------------
// featcl[B,9,9,512] (padded) conv3x3 -> polfeat[B,32,64] (p padded 49->64).
// N = 32 exactly = one 2x2 macro-tile column.
__device__ __forceinline__ void store_fin(_Float16* polfeat, int m0, int n0, v8f acc,
                                          int lane, const float* finb) {
    const int col = lane & 15, r0 = (lane >> 4) << 3;
    const float bv = finb[n0 + col];
#pragma unroll
    for (int i = 0; i < 8; ++i) {
        const int r = m0 + r0 + i;
        const int b = r / 49, p = r % 49;
        polfeat[((size_t)b * 32 + n0 + col) * 64 + p] = (_Float16)fmaxf(acc[i] + bv, 0.0f);
    }
}
__global__ void gemm_fin(const _Float16* __restrict__ featcl, const _Float16* __restrict__ Wfin,
                         const float* __restrict__ finb, _Float16* __restrict__ polfeat) {
    const int lane = threadIdx.x & 31;
    const int wpb = blockDim.x >> 5;
    const int tiles = 12544 >> 5;
    const int stride = gridDim.x * wpb;
    for (int t = blockIdx.x * wpb + (threadIdx.x >> 5); t < tiles; t += stride) {
        const int m0 = t << 5;
        const int ra0 = m0 + (lane & 15), ra1 = ra0 + 16;
        const int ba0 = ra0 / 49, pa0 = ra0 % 49, oy0 = pa0 / 7, ox0 = pa0 % 7;
        const int ba1 = ra1 / 49, pa1 = ra1 % 49, oy1 = pa1 / 7, ox1 = pa1 % 7;
        v8f a00 = {}, a01 = {}, a10 = {}, a11 = {};
        for (int kyx = 0; kyx < 9; ++kyx) {
            const int ky = kyx / 3, kx = kyx % 3;
            const _Float16* ar0 = featcl + ((size_t)ba0 * 81 + (oy0 + ky) * 9 + (ox0 + kx)) * 512;
            const _Float16* ar1 = featcl + ((size_t)ba1 * 81 + (oy1 + ky) * 9 + (ox1 + kx)) * 512;
            const _Float16* wbase = Wfin + kyx * 512;  // W layout [oc][kyx][ic], ld=4608
            for (int icb = 0; icb < 512; icb += 32) {
                v16h fa0 = load_fragA_ptr(ar0, icb, lane);
                v16h fa1 = load_fragA_ptr(ar1, icb, lane);
                v16h fb0 = load_fragB(wbase, 4608, 0, icb, lane);
                v16h fb1 = load_fragB(wbase, 4608, 16, icb, lane);
                a00 = wmma32(fa0, fb0, a00);
                a01 = wmma32(fa0, fb1, a01);
                a10 = wmma32(fa1, fb0, a10);
                a11 = wmma32(fa1, fb1, a11);
            }
        }
        store_fin(polfeat, m0,      0,  a00, lane, finb);
        store_fin(polfeat, m0,      16, a01, lane, finb);
        store_fin(polfeat, m0 + 16, 0,  a10, lane, finb);
        store_fin(polfeat, m0 + 16, 16, a11, lane, finb);
    }
}

// -------- pol GEMM: polfeat[8192,64] x pol_w[128,64]^T + b -> enc_in -------
// output row b*32+h remapped to enc_in row b*33+1+h (f16). 2x2 blocked.
__device__ __forceinline__ void store_pol(_Float16* encin, int m0, int n0, v8f acc,
                                          int lane, const float* polb) {
    const int col = lane & 15, r0 = (lane >> 4) << 3;
    const float bv = polb[n0 + col];
#pragma unroll
    for (int i = 0; i < 8; ++i) {
        const int r = m0 + r0 + i;
        const int b = r / 32, hh = r % 32;
        encin[(size_t)(b * 33 + 1 + hh) * RNNIN + n0 + col] = (_Float16)(acc[i] + bv);
    }
}
__global__ void gemm_pol(const _Float16* __restrict__ polfeat, const _Float16* __restrict__ Wp,
                         const float* __restrict__ polb, _Float16* __restrict__ encin) {
    const int lane = threadIdx.x & 31;
    const int wpb = blockDim.x >> 5;
    const int tn = RNNIN >> 5, tiles = (8192 >> 5) * tn;
    const int stride = gridDim.x * wpb;
    for (int t = blockIdx.x * wpb + (threadIdx.x >> 5); t < tiles; t += stride) {
        const int m0 = (t / tn) << 5, n0 = (t % tn) << 5;
        const _Float16* ar0 = polfeat + (size_t)(m0 + (lane & 15)) * 64;
        const _Float16* ar1 = ar0 + (size_t)16 * 64;
        v8f a00 = {}, a01 = {}, a10 = {}, a11 = {};
        for (int kb = 0; kb < 64; kb += 32) {
            v16h fa0 = load_fragA_ptr(ar0, kb, lane);
            v16h fa1 = load_fragA_ptr(ar1, kb, lane);
            v16h fb0 = load_fragB(Wp, 64, n0, kb, lane);
            v16h fb1 = load_fragB(Wp, 64, n0 + 16, kb, lane);
            a00 = wmma32(fa0, fb0, a00);
            a01 = wmma32(fa0, fb1, a01);
            a10 = wmma32(fa1, fb0, a10);
            a11 = wmma32(fa1, fb1, a11);
        }
        store_pol(encin, m0,      n0,      a00, lane, polb);
        store_pol(encin, m0,      n0 + 16, a01, lane, polb);
        store_pol(encin, m0 + 16, n0,      a10, lane, polb);
        store_pol(encin, m0 + 16, n0 + 16, a11, lane, polb);
    }
}

// ----------------------------- small kernels -------------------------------
__global__ void zero_f32(float* p, int n) {
    int i = blockIdx.x * blockDim.x + threadIdx.x;
    if (i < n) p[i] = 0.0f;
}
__global__ void zero_f16(_Float16* p, int n) {
    int i = blockIdx.x * blockDim.x + threadIdx.x;
    if (i < n) p[i] = (_Float16)0.0f;
}
// f32 -> f16 with zero-padded leading-dim (for K padding of weights)
__global__ void cvt_f32_f16(const float* __restrict__ src, _Float16* __restrict__ dst,
                            int rows, int cols, int ldd) {
    int i = blockIdx.x * blockDim.x + threadIdx.x;
    if (i >= rows * ldd) return;
    int r = i / ldd, c = i % ldd;
    dst[i] = (c < cols) ? (_Float16)src[(size_t)r * cols + c] : (_Float16)0.0f;
}
// fin_w [32,512,3,3] -> Wfin [oc][kyx][ic]  (f16)
__global__ void reorg_fin(const float* __restrict__ fw, _Float16* __restrict__ W) {
    int i = blockIdx.x * blockDim.x + threadIdx.x;
    if (i >= 32 * 9 * 512) return;
    int ic = i % 512, t = i / 512, kyx = t % 9, oc = t / 9;
    int ky = kyx / 3, kx = kyx % 3;
    W[i] = (_Float16)fw[(((size_t)oc * 512 + ic) * 3 + ky) * 3 + kx];
}
// pre conv: 4->3 ch, 3x3 pad1, input scaled 1e6, f16 out [B,3,32,64]
__global__ void preconv(const float* __restrict__ cm, const float* __restrict__ pw,
                        const float* __restrict__ pb, _Float16* __restrict__ x) {
    int idx = blockIdx.x * blockDim.x + threadIdx.x;
    if (idx >= B_ * 3 * 32 * 64) return;
    int xw = idx & 63, yy = (idx >> 6) & 31, t = idx >> 11, oc = t % 3, b = t / 3;
    float s = pb[oc];
    for (int ic = 0; ic < 4; ++ic)
        for (int ky = 0; ky < 3; ++ky) {
            int iy = yy - 1 + ky; if (iy < 0 || iy >= 32) continue;
            for (int kx = 0; kx < 3; ++kx) {
                int ix = xw - 1 + kx; if (ix < 0 || ix >= 64) continue;
                s += cm[(((size_t)b * 4 + ic) * 32 + iy) * 64 + ix] * 1e6f *
                     pw[((oc * 4 + ic) * 3 + ky) * 3 + kx];
            }
        }
    x[idx] = (_Float16)s;
}
// im2col for stride-32 res conv on zero-padded 224x224 canvas (only 32x64 live)
__global__ void im2col_res(const _Float16* __restrict__ x, _Float16* __restrict__ Ares) {
    int idx = blockIdx.x * blockDim.x + threadIdx.x;
    if (idx >= 12544 * 32) return;
    int k = idx & 31, m = idx >> 5;
    _Float16 v = (_Float16)0.0f;
    if (k < 27) {
        int b = m / 49, p = m % 49, oy = p / 7, ox = p % 7;
        int ic = k / 9, t = k % 9, ky = t / 3, kx = t % 3;
        int iy = 32 * oy - 1 + ky, ix = 32 * ox - 1 + kx;
        if (iy >= 0 && iy < 32 && ix >= 0 && ix < 64)
            v = x[(((size_t)b * 3 + ic) * 32 + iy) * 64 + ix];
    }
    Ares[idx] = v;
}
// value head: mean over 7x7 then dot val_w
__global__ void value_head(const _Float16* __restrict__ featcl, const float* __restrict__ vw,
                           const float* __restrict__ vb, float* __restrict__ out) {
    int b = blockIdx.x, tid = threadIdx.x;
    float s = 0.0f;
    for (int oc = tid; oc < 512; oc += 256) {
        float a = 0.0f;
        for (int p = 0; p < 49; ++p) {
            int oy = p / 7, ox = p % 7;
            a += (float)featcl[((size_t)b * 81 + (oy + 1) * 9 + (ox + 1)) * 512 + oc];
        }
        s += a * vw[oc];
    }
    __shared__ float red[256];
    red[tid] = s; __syncthreads();
    for (int o = 128; o > 0; o >>= 1) { if (tid < o) red[tid] += red[tid + o]; __syncthreads(); }
    if (tid == 0) out[b] = red[0] / 49.0f + vb[0];
}
// bos rows + decoder init
__global__ void init_seq(const float* __restrict__ emb, _Float16* __restrict__ encin,
                         _Float16* __restrict__ xdec, int* __restrict__ prev) {
    int idx = blockIdx.x * blockDim.x + threadIdx.x;
    if (idx < B_ * RNNIN) {
        int b = idx >> 7, w = idx & 127;
        encin[(size_t)(b * 33) * RNNIN + w] = (_Float16)emb[w];
        xdec[idx] = (_Float16)emb[RNNIN + w];
    }
    if (idx < B_) prev[idx] = -1;
}
__device__ __forceinline__ float sigm(float x) { return 1.0f / (1.0f + expf(-x)); }
// GRU gate fuse: h' = (1-z)*n + z*h   (gi/gh already include biases)
__global__ void gru_gate(const float* __restrict__ gi_base, int gi_stride,
                         const float* __restrict__ gh,
                         float* __restrict__ h, _Float16* __restrict__ h16,
                         _Float16* __restrict__ enc_out, int eo_stride) {
    int idx = blockIdx.x * blockDim.x + threadIdx.x;
    if (idx >= B_ * HID_) return;
    int b = idx >> 8, u = idx & 255;
    const float* gi = gi_base + (size_t)b * gi_stride;
    const float* gr = gh + (size_t)b * G3;
    float r = sigm(gi[u] + gr[u]);
    float z = sigm(gi[HID_ + u] + gr[HID_ + u]);
    float n = tanhf(gi[2 * HID_ + u] + r * gr[2 * HID_ + u]);
    float hn = (1.0f - z) * n + z * h[idx];
    h[idx] = hn;
    h16[idx] = (_Float16)hn;
    if (enc_out) enc_out[(size_t)b * eo_stride + u] = (_Float16)hn;
}
__device__ __forceinline__ unsigned mix32(unsigned x) {
    x ^= x >> 16; x *= 0x7feb352dU; x ^= x >> 15; x *= 0x846ca68bU; x ^= x >> 16;
    return x;
}
// attention + Gumbel-max categorical + carry update (one block per batch)
__global__ void attn_sample(const float* __restrict__ KeyT, const float* __restrict__ q,
                            const float* __restrict__ Wv,
                            float* __restrict__ mask, float* __restrict__ logp_sum,
                            int* __restrict__ prev,
                            const _Float16* __restrict__ encin, _Float16* __restrict__ xdec,
                            int step) {
    int b = blockIdx.x, tid = threadIdx.x;  // 64 threads
    __shared__ float logits[LSEQ];
    __shared__ int samp_sh;
    if (tid < LSEQ) {
        const float* kr = KeyT + (size_t)(b * LSEQ + tid) * WD_;
        const float* qr = q + (size_t)b * WD_;
        float s = 0.0f;
        for (int w = 0; w < WD_; ++w) s += Wv[w] * tanhf(kr[w] + qr[w]);
        logits[tid] = fmaxf(s, 0.0f) - 1e7f * mask[b * LSEQ + tid];
    }
    __syncthreads();
    if (tid == 0) {
        float mx = -3.0e38f;
        for (int l = 0; l < LSEQ; ++l) mx = fmaxf(mx, logits[l]);
        float se = 0.0f;
        for (int l = 0; l < LSEQ; ++l) se += expf(logits[l] - mx);
        float lse = mx + logf(se);
        int samp = 0; float best = -3.0e38f;
        for (int l = 0; l < LSEQ; ++l) {
            unsigned h = mix32(0x2A6A195Fu ^ mix32((unsigned)step * 0x9E3779B9u ^
                               (unsigned)b * 0x85EBCA6Bu ^ (unsigned)l * 0xC2B2AE35u));
            float u = ((float)((h >> 8) + 1u)) * (1.0f / 16777218.0f);
            float g = -logf(-logf(u));
            if (logits[l] + g > best) { best = logits[l] + g; samp = l; }
        }
        bool terminal = (prev[b] == 0);
        if (terminal || step == MAXS_) samp = 0;
        float logp = logits[samp] - lse;
        logp_sum[b] += logp * (terminal ? 0.0f : 1.0f);
        mask[b * LSEQ + samp] = 1.0f;
        prev[b] = samp;
        samp_sh = samp;
    }
    __syncthreads();
    int samp = samp_sh;
    for (int w = tid; w < RNNIN; w += blockDim.x)
        xdec[(size_t)b * RNNIN + w] = encin[(size_t)(b * LSEQ + samp) * RNNIN + w];
}

// ------------------------------- host side ---------------------------------
static inline int gblk(int tiles) { int bl = (tiles + 7) / 8; return bl > 6144 ? 6144 : bl; }

extern "C" void kernel_launch(void* const* d_in, const int* in_sizes, int n_in,
                              void* d_out, int out_size, void* d_ws, size_t ws_size,
                              hipStream_t stream) {
    (void)in_sizes; (void)n_in; (void)out_size; (void)ws_size;
    const float* cm      = (const float*)d_in[0];
    const float* pre_w   = (const float*)d_in[1];
    const float* pre_b   = (const float*)d_in[2];
    const float* res_w   = (const float*)d_in[3];
    const float* res_b   = (const float*)d_in[4];
    const float* fin_w   = (const float*)d_in[5];
    const float* fin_b   = (const float*)d_in[6];
    const float* emb     = (const float*)d_in[7];
    const float* enc_wih = (const float*)d_in[8];
    const float* enc_whh = (const float*)d_in[9];
    const float* enc_bih = (const float*)d_in[10];
    const float* enc_bhh = (const float*)d_in[11];
    const float* dec_wih = (const float*)d_in[12];
    const float* dec_whh = (const float*)d_in[13];
    const float* dec_bih = (const float*)d_in[14];
    const float* dec_bhh = (const float*)d_in[15];
    const float* W_k     = (const float*)d_in[16];
    const float* W_q     = (const float*)d_in[17];
    const float* W_v     = (const float*)d_in[18];
    const float* val_w   = (const float*)d_in[19];
    const float* val_b   = (const float*)d_in[20];
    const float* pol_w   = (const float*)d_in[21];
    const float* pol_b   = (const float*)d_in[22];

    float* out = (float*)d_out;
    float* out_logp  = out;            // [256]
    float* out_mask  = out + 256;      // [256*33]
    float* out_value = out + 8704;     // [256]

    // ---- workspace carve-out (256B aligned bump allocator) ----
    char* wsp = (char*)d_ws;
    auto alloc = [&](size_t bytes) -> char* {
        char* p = wsp; wsp += (bytes + 255) & ~(size_t)255; return p;
    };
    _Float16* x16      = (_Float16*)alloc((size_t)B_ * 3 * 32 * 64 * 2);
    _Float16* Ares     = (_Float16*)alloc((size_t)12544 * 32 * 2);
    _Float16* Wres16   = (_Float16*)alloc((size_t)512 * 32 * 2);
    _Float16* featcl   = (_Float16*)alloc((size_t)B_ * 81 * 512 * 2);
    _Float16* Wfin16   = (_Float16*)alloc((size_t)32 * 9 * 512 * 2);
    _Float16* polfeat  = (_Float16*)alloc((size_t)8192 * 64 * 2);
    _Float16* Wpol16   = (_Float16*)alloc((size_t)128 * 64 * 2);
    _Float16* encin    = (_Float16*)alloc((size_t)8448 * RNNIN * 2);
    _Float16* Weih16   = (_Float16*)alloc((size_t)G3 * RNNIN * 2);
    _Float16* Wehh16   = (_Float16*)alloc((size_t)G3 * HID_ * 2);
    _Float16* Wdih16   = (_Float16*)alloc((size_t)G3 * RNNIN * 2);
    _Float16* Wdhh16   = (_Float16*)alloc((size_t)G3 * HID_ * 2);
    _Float16* Wk16     = (_Float16*)alloc((size_t)WD_ * HID_ * 2);
    _Float16* Wq16     = (_Float16*)alloc((size_t)WD_ * HID_ * 2);
    _Float16* h16      = (_Float16*)alloc((size_t)B_ * HID_ * 2);
    _Float16* xdec     = (_Float16*)alloc((size_t)B_ * RNNIN * 2);
    _Float16* encout16 = (_Float16*)alloc((size_t)8448 * HID_ * 2);
    float*    gi_enc   = (float*)alloc((size_t)8448 * G3 * 4);
    float*    gh       = (float*)alloc((size_t)B_ * G3 * 4);
    float*    gi_d     = (float*)alloc((size_t)B_ * G3 * 4);
    float*    hbuf     = (float*)alloc((size_t)B_ * HID_ * 4);
    float*    KeyT     = (float*)alloc((size_t)8448 * WD_ * 4);
    float*    qbuf     = (float*)alloc((size_t)B_ * WD_ * 4);
    int*      prev     = (int*)alloc((size_t)B_ * 4);

    // ---- init / weight conversion (every call: deterministic) ----
    zero_f32<<<(8960 + 255) / 256, 256, 0, stream>>>(out, 8960);
    zero_f32<<<(B_ * HID_ + 255) / 256, 256, 0, stream>>>(hbuf, B_ * HID_);
    zero_f16<<<(B_ * HID_ + 255) / 256, 256, 0, stream>>>(h16, B_ * HID_);
    {
        int n = B_ * 81 * 512;
        zero_f16<<<(n + 255) / 256, 256, 0, stream>>>(featcl, n);
        n = 8192 * 64;
        zero_f16<<<(n + 255) / 256, 256, 0, stream>>>(polfeat, n);
    }
    cvt_f32_f16<<<(G3 * RNNIN + 255) / 256, 256, 0, stream>>>(enc_wih, Weih16, G3, RNNIN, RNNIN);
    cvt_f32_f16<<<(G3 * HID_ + 255) / 256, 256, 0, stream>>>(enc_whh, Wehh16, G3, HID_, HID_);
    cvt_f32_f16<<<(G3 * RNNIN + 255) / 256, 256, 0, stream>>>(dec_wih, Wdih16, G3, RNNIN, RNNIN);
    cvt_f32_f16<<<(G3 * HID_ + 255) / 256, 256, 0, stream>>>(dec_whh, Wdhh16, G3, HID_, HID_);
    cvt_f32_f16<<<(WD_ * HID_ + 255) / 256, 256, 0, stream>>>(W_k, Wk16, WD_, HID_, HID_);
    cvt_f32_f16<<<(WD_ * HID_ + 255) / 256, 256, 0, stream>>>(W_q, Wq16, WD_, HID_, HID_);
    cvt_f32_f16<<<(128 * 64 + 255) / 256, 256, 0, stream>>>(pol_w, Wpol16, 128, 49, 64);
    cvt_f32_f16<<<(512 * 32 + 255) / 256, 256, 0, stream>>>(res_w, Wres16, 512, 27, 32);
    reorg_fin<<<(32 * 9 * 512 + 255) / 256, 256, 0, stream>>>(fin_w, Wfin16);

    // ---- CNN trunk ----
    preconv<<<(B_ * 3 * 32 * 64 + 255) / 256, 256, 0, stream>>>(cm, pre_w, pre_b, x16);
    im2col_res<<<(12544 * 32 + 255) / 256, 256, 0, stream>>>(x16, Ares);
    gemm_res<<<gblk(392 * 16), 256, 0, stream>>>(Ares, Wres16, res_b, featcl);
    value_head<<<B_, 256, 0, stream>>>(featcl, val_w, val_b, out_value);
    gemm_fin<<<gblk(392), 256, 0, stream>>>(featcl, Wfin16, fin_b, polfeat);
    gemm_pol<<<gblk(256 * 4), 256, 0, stream>>>(polfeat, Wpol16, pol_b, encin);
    init_seq<<<(B_ * RNNIN + 255) / 256, 256, 0, stream>>>(emb, encin, xdec, prev);

    // ---- encoder GRU: batched input GEMM + 33 recurrent steps ----
    gemm_f16f32<<<gblk((8448 / 32) * (G3 / 32)), 256, 0, stream>>>(
        encin, RNNIN, Weih16, RNNIN, gi_enc, G3, 8448, G3, RNNIN, enc_bih, 0);
    for (int t = 0; t < LSEQ; ++t) {
        gemm_f16f32<<<gblk((B_ / 32) * (G3 / 32)), 256, 0, stream>>>(
            h16, HID_, Wehh16, HID_, gh, G3, B_, G3, HID_, enc_bhh, 0);
        gru_gate<<<(B_ * HID_ + 255) / 256, 256, 0, stream>>>(
            gi_enc + (size_t)t * G3, LSEQ * G3, gh, hbuf, h16,
            encout16 + (size_t)t * HID_, LSEQ * HID_);
    }
    // Key^T[b*33+l, w] = enc_out . W_k
    gemm_f16f32<<<gblk((8448 / 32) * (WD_ / 32)), 256, 0, stream>>>(
        encout16, HID_, Wk16, HID_, KeyT, WD_, 8448, WD_, HID_, nullptr, 0);

    // ---- decoder: 17 steps ----
    for (int t = 0; t <= MAXS_; ++t) {
        gemm_f16f32<<<gblk((B_ / 32) * (G3 / 32)), 256, 0, stream>>>(
            xdec, RNNIN, Wdih16, RNNIN, gi_d, G3, B_, G3, RNNIN, dec_bih, 0);
        gemm_f16f32<<<gblk((B_ / 32) * (G3 / 32)), 256, 0, stream>>>(
            h16, HID_, Wdhh16, HID_, gh, G3, B_, G3, HID_, dec_bhh, 0);
        gru_gate<<<(B_ * HID_ + 255) / 256, 256, 0, stream>>>(
            gi_d, G3, gh, hbuf, h16, (_Float16*)nullptr, 0);
        gemm_f16f32<<<gblk((B_ / 32) * (WD_ / 32)), 256, 0, stream>>>(
            h16, HID_, Wq16, HID_, qbuf, WD_, B_, WD_, HID_, nullptr, 0);
        attn_sample<<<B_, 64, 0, stream>>>(KeyT, qbuf, W_v, out_mask, out_logp,
                                           prev, encin, xdec, t);
    }
}
